// EncoderME_19774029431201
// MI455X (gfx1250) — compile-verified, matrix operand order checked by reading, not run on hardware
//
#include <hip/hip_runtime.h>
#include <hip/hip_bf16.h>
#include <stdint.h>

// ---------------------------------------------------------------------------
// Types for CDNA5 WMMA (wave32, 16x16x32 bf16 -> f32)
// ---------------------------------------------------------------------------
typedef __attribute__((ext_vector_type(16))) __bf16 v16bf;
typedef __attribute__((ext_vector_type(8)))  float  v8f;

union Frag { uint4 q[2]; v16bf v; };

__device__ __forceinline__ unsigned short f2bf(float f) {
  unsigned u = __float_as_uint(f);
  u += 0x7fffu + ((u >> 16) & 1u);          // round-to-nearest-even
  return (unsigned short)(u >> 16);
}
__device__ __forceinline__ float bf2f(unsigned short h) {
  return __uint_as_float(((unsigned)h) << 16);
}

// ---------------------------------------------------------------------------
// Implicit-GEMM conv via WMMA, 2x(NB) register-blocked:
// one wave computes a 32(pixel) x 16*NB(cout) tile with 2*NB accumulators,
// so each K-step issues 4 + 2*NB b128 loads and 2*NB back-to-back WMMAs.
// Activations: NHWC bf16. Weights pre-packed: [tap][cout][cin] bf16.
// MODE 0: write raw bf16 conv output + accumulate per-channel BN stats
// MODE 1: add bias, ReLU, write bf16    (linear-1 head)
// MODE 2: add bias, write fp32          (linear-2 head -> d_out)
// PADK  : Cin==16, upper K-half of each 32-chunk must be zero-padded
// NOPAD : input coords provably in bounds (2x2 s2 VALID, 1x1) -> no selects
// ---------------------------------------------------------------------------
template<int MODE, int NB, bool PADK, bool NOPAD>
__global__ __launch_bounds__(256)
void conv_wmma_kernel(
    const unsigned short* __restrict__ act,
    const unsigned short* __restrict__ wpk,
    float* __restrict__ sum, float* __restrict__ sumsq,
    unsigned short* __restrict__ out_bf, float* __restrict__ out_f32,
    const float* __restrict__ bias,
    int B, int IH, int IW, int Cin, int OH, int OW, int Cout,
    int KH, int KW, int SH, int SW, int PAD)
{
  const int wave = (int)((blockIdx.x * blockDim.x + threadIdx.x) >> 5);
  const int lane = threadIdx.x & 31;
  const int P = B * OH * OW;                 // pixels (multiple of 32)
  const int nBlocks = Cout / (16 * NB);
  const int mBlocks = P >> 5;
  if (wave >= mBlocks * nBlocks) return;     // uniform per wave
  const int mt = wave / nBlocks, nt = wave % nBlocks;
  const int m0 = mt << 5, n0 = nt * (16 * NB);
  const int row  = lane & 15;                // M index for A, N index for B/C
  const int half = lane >> 4;                // K-half selector

  int ohs[2], ows[2], bs[2];
#pragma unroll
  for (int s = 0; s < 2; ++s) {
    const int p = m0 + 16 * s + row;
    bs[s] = p / (OH * OW);
    const int r = p % (OH * OW);
    ohs[s] = r / OW; ows[s] = r % OW;
  }

  const v8f vzero = {0.f, 0.f, 0.f, 0.f, 0.f, 0.f, 0.f, 0.f};
  v8f acc[2][NB];
#pragma unroll
  for (int s = 0; s < 2; ++s)
#pragma unroll
    for (int t = 0; t < NB; ++t) acc[s][t] = vzero;

  const uint4 z4 = make_uint4(0u, 0u, 0u, 0u);

  for (int kh = 0; kh < KH; ++kh) {
    for (int kw = 0; kw < KW; ++kw) {
      const unsigned short* abase[2];
      bool inb[2];
#pragma unroll
      for (int s = 0; s < 2; ++s) {
        const int ih = ohs[s] * SH + kh - PAD;
        const int iw = ows[s] * SW + kw - PAD;
        inb[s] = NOPAD ? true
                       : (((unsigned)ih < (unsigned)IH) &&
                          ((unsigned)iw < (unsigned)IW));
        abase[s] = act + ((size_t)(bs[s] * IH + ih) * IW + iw) * (size_t)Cin;
        if (inb[s]) __builtin_prefetch((const void*)abase[s], 0, 0);
      }
      const unsigned short* wrow[NB];
#pragma unroll
      for (int t = 0; t < NB; ++t)
        wrow[t] = wpk + ((size_t)(kh * KW + kw) * Cout +
                         (n0 + 16 * t + row)) * (size_t)Cin;

      for (int cc = 0; cc < Cin; cc += 32) {
        // 16-bit A layout: lane-half h holds K = cc+8h..+7 (elems 0..7)
        // and K = cc+16+8h..+7 (elems 8..15) -> two contiguous 16B chunks.
        const int cLo = cc + 8 * half;
        const int cHi = cc + 16 + 8 * half;
        Frag A[2], Bf[NB];
#pragma unroll
        for (int s = 0; s < 2; ++s) {
          const bool okLo = PADK ? (inb[s] && (cLo + 8 <= Cin)) : inb[s];
          const bool okHi = PADK ? (inb[s] && (cHi + 8 <= Cin)) : inb[s];
          A[s].q[0] = okLo ? *(const uint4*)(abase[s] + cLo) : z4;
          A[s].q[1] = okHi ? *(const uint4*)(abase[s] + cHi) : z4;
        }
#pragma unroll
        for (int t = 0; t < NB; ++t) {
          Bf[t].q[0] = (!PADK || (cLo + 8 <= Cin)) ? *(const uint4*)(wrow[t] + cLo) : z4;
          Bf[t].q[1] = (!PADK || (cHi + 8 <= Cin)) ? *(const uint4*)(wrow[t] + cHi) : z4;
        }
#pragma unroll
        for (int s = 0; s < 2; ++s)
#pragma unroll
          for (int t = 0; t < NB; ++t)
            acc[s][t] = __builtin_amdgcn_wmma_f32_16x16x32_bf16(
                            false, A[s].v, false, Bf[t].v,
                            (short)0, acc[s][t], false, false);
      }
    }
  }

  // C/D layout: VGPR v -> (M = v + 8*half, N = row)
  if (MODE == 0) {
#pragma unroll
    for (int t = 0; t < NB; ++t) {
      const int ch = n0 + 16 * t + row;
      float s1 = 0.f, s2 = 0.f;
#pragma unroll
      for (int s = 0; s < 2; ++s) {
#pragma unroll
        for (int v = 0; v < 8; ++v) {
          const float val = acc[s][t][v];
          const int pOut = m0 + 16 * s + v + 8 * half;
          out_bf[(size_t)pOut * Cout + ch] = f2bf(val);
          s1 += val; s2 += val * val;
        }
      }
      atomicAdd(&sum[ch], s1);
      atomicAdd(&sumsq[ch], s2);
    }
  } else if (MODE == 1) {
#pragma unroll
    for (int t = 0; t < NB; ++t) {
      const int ch = n0 + 16 * t + row;
      const float bv = bias[ch];
#pragma unroll
      for (int s = 0; s < 2; ++s) {
#pragma unroll
        for (int v = 0; v < 8; ++v) {
          float val = acc[s][t][v] + bv;
          val = val > 0.f ? val : 0.f;
          const int pOut = m0 + 16 * s + v + 8 * half;
          out_bf[(size_t)pOut * Cout + ch] = f2bf(val);
        }
      }
    }
  } else {
#pragma unroll
    for (int t = 0; t < NB; ++t) {
      const int ch = n0 + 16 * t + row;
      const float bv = bias[ch];
#pragma unroll
      for (int s = 0; s < 2; ++s) {
#pragma unroll
        for (int v = 0; v < 8; ++v) {
          const int pOut = m0 + 16 * s + v + 8 * half;
          out_f32[(size_t)pOut * Cout + ch] = acc[s][t][v] + bv;
        }
      }
    }
  }
}

// ---------------------------------------------------------------------------
// Stem: 2x1 stride-(2,1) conv, cin=1 -> 16, fp32 in, bf16 raw out + BN stats
// ---------------------------------------------------------------------------
__global__ __launch_bounds__(256)
void layer0_kernel(const float* __restrict__ x, const float* __restrict__ w,
                   unsigned short* __restrict__ raw,
                   float* __restrict__ sum, float* __restrict__ sumsq,
                   int B, int IH, int IW, int OH, int OW)
{
  __shared__ float ss[16], sq[16];
  if (threadIdx.x < 16) { ss[threadIdx.x] = 0.f; sq[threadIdx.x] = 0.f; }
  __syncthreads();
  const int p = blockIdx.x * blockDim.x + threadIdx.x;
  const int P = B * OH * OW;
  if (p < P) {
    const int b  = p / (OH * OW);
    const int r  = p % (OH * OW);
    const int oh = r / OW, ow = r % OW;
    const float x0 = x[((size_t)b * IH + 2 * oh    ) * IW + ow];
    const float x1 = x[((size_t)b * IH + 2 * oh + 1) * IW + ow];
    unsigned short o[16] __attribute__((aligned(16)));
#pragma unroll
    for (int ci = 0; ci < 16; ++ci) {
      const float v = x0 * w[ci] + x1 * w[16 + ci];   // HWIO (2,1,1,16)
      o[ci] = f2bf(v);
      atomicAdd(&ss[ci], v);
      atomicAdd(&sq[ci], v * v);
    }
    uint4* dst = (uint4*)(raw + (size_t)p * 16);
    dst[0] = ((const uint4*)o)[0];
    dst[1] = ((const uint4*)o)[1];
  }
  __syncthreads();
  if (threadIdx.x < 16) {
    atomicAdd(&sum[threadIdx.x],   ss[threadIdx.x]);
    atomicAdd(&sumsq[threadIdx.x], sq[threadIdx.x]);
  }
}

// ---------------------------------------------------------------------------
// BN finalize: scale = g*rsqrt(var+eps), shift = b - mean*scale
// ---------------------------------------------------------------------------
__global__ void bnfin_kernel(const float* __restrict__ sum, const float* __restrict__ sumsq,
                             const float* __restrict__ g, const float* __restrict__ bb,
                             float* __restrict__ scale, float* __restrict__ shift,
                             int C, float inv)
{
  const int ci = blockIdx.x * blockDim.x + threadIdx.x;
  if (ci >= C) return;
  const float mean = sum[ci] * inv;
  const float var  = sumsq[ci] * inv - mean * mean;
  const float rs   = rsqrtf(var + 1e-5f);
  const float sc   = g[ci] * rs;
  scale[ci] = sc;
  shift[ci] = bb[ci] - mean * sc;
}

// ---------------------------------------------------------------------------
// Fused BN-apply + ReLU, bf16 -> bf16, 8 channels per thread
// ---------------------------------------------------------------------------
__global__ __launch_bounds__(256)
void bnrelu_kernel(const unsigned short* __restrict__ raw,
                   const float* __restrict__ scale, const float* __restrict__ shift,
                   unsigned short* __restrict__ out, int C, long n8)
{
  const long i = (long)blockIdx.x * blockDim.x + threadIdx.x;
  if (i >= n8) return;
  const long base = i * 8;
  const int ch0 = (int)(base % C);          // C is a multiple of 8
  uint4 q = *(const uint4*)(raw + base);
  const unsigned short* s = (const unsigned short*)&q;
  unsigned short o[8] __attribute__((aligned(16)));
#pragma unroll
  for (int j = 0; j < 8; ++j) {
    float y = bf2f(s[j]) * scale[ch0 + j] + shift[ch0 + j];
    o[j] = f2bf(y > 0.f ? y : 0.f);
  }
  *(uint4*)(out + base) = *(const uint4*)o;
}

// ---------------------------------------------------------------------------
// Helpers: zero stats, pack HWIO fp32 -> [tap][cout][cin] bf16, plain convert
// ---------------------------------------------------------------------------
__global__ void zero_f32_kernel(float* __restrict__ p, int n) {
  const int i = blockIdx.x * blockDim.x + threadIdx.x;
  if (i < n) p[i] = 0.f;
}

__global__ void pack_conv_w_kernel(const float* __restrict__ w, unsigned short* __restrict__ out,
                                   int taps, int Cin, int Cout)
{
  const int i = blockIdx.x * blockDim.x + threadIdx.x;
  const int n = taps * Cin * Cout;
  if (i >= n) return;
  const int co  = i % Cout;
  const int t2  = i / Cout;
  const int ci  = t2 % Cin;
  const int tap = t2 / Cin;
  out[((size_t)tap * Cout + co) * Cin + ci] = f2bf(w[i]);
}

__global__ void f32_to_bf16_kernel(const float* __restrict__ w, unsigned short* __restrict__ out, int n) {
  const int i = blockIdx.x * blockDim.x + threadIdx.x;
  if (i < n) out[i] = f2bf(w[i]);
}

// ---------------------------------------------------------------------------
// Launcher
// ---------------------------------------------------------------------------
extern "C" void kernel_launch(void* const* d_in, const int* in_sizes, int n_in,
                              void* d_out, int out_size, void* d_ws, size_t ws_size,
                              hipStream_t stream) {
  (void)in_sizes; (void)n_in; (void)out_size;
  const int B = 16;

  // ---- inputs (setup_inputs dict order, lists flattened in order) ----
  const float* x = (const float*)d_in[0];
  const float* conv_w[15]; const float* bn_g[15]; const float* bn_b[15];
  for (int i = 0; i < 15; ++i) {
    conv_w[i] = (const float*)d_in[1 + i];
    bn_g[i]   = (const float*)d_in[16 + i];
    bn_b[i]   = (const float*)d_in[31 + i];
  }
  const float* lin_w1 = (const float*)d_in[46];
  const float* lin_b1 = (const float*)d_in[47];
  const float* lin_w2 = (const float*)d_in[48];
  const float* lin_b2 = (const float*)d_in[49];

  // ---- layer configs ----
  int KHs[15], KWs[15], SHs[15], SWs[15], PADs[15];
  int Cins[15], Couts[15], IHs[15], IWs[15], OHs[15], OWs[15];
  KHs[0]=2; KWs[0]=1; SHs[0]=2; SWs[0]=1; PADs[0]=0;
  Cins[0]=1; Couts[0]=16; IHs[0]=512; IWs[0]=256; OHs[0]=256; OWs[0]=256;
  {
    int H = 256, W = 256, C = 16;
    for (int i = 0; i < 7; ++i) {
      int L = 1 + 2 * i;   // 3x3 SAME
      KHs[L]=3; KWs[L]=3; SHs[L]=1; SWs[L]=1; PADs[L]=1;
      Cins[L]=C; Couts[L]=C; IHs[L]=H; IWs[L]=W; OHs[L]=H; OWs[L]=W;
      L = 2 + 2 * i;       // 2x2 stride-2 VALID
      KHs[L]=2; KWs[L]=2; SHs[L]=2; SWs[L]=2; PADs[L]=0;
      Cins[L]=C; Couts[L]=2*C; IHs[L]=H; IWs[L]=W; OHs[L]=H/2; OWs[L]=W/2;
      H /= 2; W /= 2; C *= 2;
    }
  }

  // ---- workspace layout ----
  const int TOTAL_CH = 6112;                      // sum of Cout over 15 layers
  const size_t STATS_BYTES = 98304;               // 4 * 6112 * 4B, rounded
  size_t wpkElems = 0;
  for (int L = 1; L < 15; ++L) wpkElems += (size_t)KHs[L]*KWs[L]*Cins[L]*Couts[L];
  const size_t lin1Elems = (size_t)2048 * 256;
  const size_t lin2Elems = (size_t)256 * 128;
  const size_t WPK_ELEMS = wpkElems + lin1Elems + lin2Elems;   // 24,323,328
  const size_t ACT = (size_t)16 * 256 * 256 * 16;              // max activation elems
  const size_t REQUIRED = STATS_BYTES + 2 * WPK_ELEMS + 3 * 2 * ACT;
  if (ws_size < REQUIRED) return;

  char* wsb = (char*)d_ws;
  float* sumBase   = (float*)wsb;
  float* sumsqBase = sumBase + TOTAL_CH;
  float* scaleBase = sumBase + 2 * TOTAL_CH;
  float* shiftBase = sumBase + 3 * TOTAL_CH;
  unsigned short* wpkBase = (unsigned short*)(wsb + STATS_BYTES);
  unsigned short* b0 = (unsigned short*)(wsb + STATS_BYTES + 2 * WPK_ELEMS);
  unsigned short* b1 = b0 + ACT;
  unsigned short* b2 = b1 + ACT;

  auto cdiv = [](long a, long b) -> int { return (int)((a + b - 1) / b); };

  // ---- zero BN accumulators (deterministic every call) ----
  zero_f32_kernel<<<cdiv(2 * TOTAL_CH, 256), 256, 0, stream>>>(sumBase, 2 * TOTAL_CH);

  // ---- pack weights to bf16 [tap][cout][cin] ----
  size_t wpkOff[15]; wpkOff[0] = 0;
  {
    size_t o = 0;
    for (int L = 1; L < 15; ++L) {
      wpkOff[L] = o;
      const int taps = KHs[L] * KWs[L];
      const int n = taps * Cins[L] * Couts[L];
      pack_conv_w_kernel<<<cdiv(n, 256), 256, 0, stream>>>(
          conv_w[L], wpkBase + o, taps, Cins[L], Couts[L]);
      o += (size_t)n;
    }
    // linear weights are already [cout][cin]; straight convert
    f32_to_bf16_kernel<<<cdiv((long)lin1Elems, 256), 256, 0, stream>>>(
        lin_w1, wpkBase + wpkElems, (int)lin1Elems);
    f32_to_bf16_kernel<<<cdiv((long)lin2Elems, 256), 256, 0, stream>>>(
        lin_w2, wpkBase + wpkElems + lin1Elems, (int)lin2Elems);
  }

  // ---- per-layer BN stat offsets ----
  int chOff[15]; chOff[0] = 0;
  for (int L = 1; L < 15; ++L) chOff[L] = chOff[L - 1] + Couts[L - 1];

  // ---- layer 0 (stem) ----
  {
    const int P = B * OHs[0] * OWs[0];
    layer0_kernel<<<cdiv(P, 256), 256, 0, stream>>>(
        x, conv_w[0], b0, sumBase + chOff[0], sumsqBase + chOff[0],
        B, IHs[0], IWs[0], OHs[0], OWs[0]);
    bnfin_kernel<<<1, 64, 0, stream>>>(
        sumBase + chOff[0], sumsqBase + chOff[0], bn_g[0], bn_b[0],
        scaleBase + chOff[0], shiftBase + chOff[0], Couts[0], 1.f / (float)P);
    const long n8 = (long)P * Couts[0] / 8;
    bnrelu_kernel<<<cdiv(n8, 256), 256, 0, stream>>>(
        b0, scaleBase + chOff[0], shiftBase + chOff[0], b1, Couts[0], n8);
  }

  // ---- layers 1..14: WMMA conv -> BN finalize -> BN+ReLU ----
  unsigned short* act_in = b1;                       // L0 output
  for (int L = 1; L < 15; ++L) {
    unsigned short* raw = b0;
    unsigned short* act_out = (L & 1) ? b2 : b1;
    const int P = B * OHs[L] * OWs[L];
    const int NBv = (Couts[L] % 32 == 0) ? 2 : 1;
    const long waves = ((long)P >> 5) * (Couts[L] / (16 * NBv));
    const dim3 grid(cdiv(waves * 32, 256));
    if (NBv == 1) {
      // L1: 3x3, Cin=16, Cout=16
      conv_wmma_kernel<0, 1, true, false><<<grid, 256, 0, stream>>>(
          act_in, wpkBase + wpkOff[L],
          sumBase + chOff[L], sumsqBase + chOff[L], raw, nullptr, nullptr,
          B, IHs[L], IWs[L], Cins[L], OHs[L], OWs[L], Couts[L],
          KHs[L], KWs[L], SHs[L], SWs[L], PADs[L]);
    } else if (Cins[L] == 16) {
      // L2: 2x2 s2 VALID (no padding), Cin=16
      conv_wmma_kernel<0, 2, true, true><<<grid, 256, 0, stream>>>(
          act_in, wpkBase + wpkOff[L],
          sumBase + chOff[L], sumsqBase + chOff[L], raw, nullptr, nullptr,
          B, IHs[L], IWs[L], Cins[L], OHs[L], OWs[L], Couts[L],
          KHs[L], KWs[L], SHs[L], SWs[L], PADs[L]);
    } else if (PADs[L] == 1) {
      // 3x3 SAME, big Cin: edge selects needed, K-guards elided
      conv_wmma_kernel<0, 2, false, false><<<grid, 256, 0, stream>>>(
          act_in, wpkBase + wpkOff[L],
          sumBase + chOff[L], sumsqBase + chOff[L], raw, nullptr, nullptr,
          B, IHs[L], IWs[L], Cins[L], OHs[L], OWs[L], Couts[L],
          KHs[L], KWs[L], SHs[L], SWs[L], PADs[L]);
    } else {
      // 2x2 s2 VALID, big Cin: fully branch-free inner loop
      conv_wmma_kernel<0, 2, false, true><<<grid, 256, 0, stream>>>(
          act_in, wpkBase + wpkOff[L],
          sumBase + chOff[L], sumsqBase + chOff[L], raw, nullptr, nullptr,
          B, IHs[L], IWs[L], Cins[L], OHs[L], OWs[L], Couts[L],
          KHs[L], KWs[L], SHs[L], SWs[L], PADs[L]);
    }
    bnfin_kernel<<<cdiv(Couts[L], 256), 256, 0, stream>>>(
        sumBase + chOff[L], sumsqBase + chOff[L], bn_g[L], bn_b[L],
        scaleBase + chOff[L], shiftBase + chOff[L], Couts[L], 1.f / (float)P);
    const long n8 = (long)P * Couts[L] / 8;
    bnrelu_kernel<<<cdiv(n8, 256), 256, 0, stream>>>(
        raw, scaleBase + chOff[L], shiftBase + chOff[L], act_out, Couts[L], n8);
    act_in = act_out;
  }

  // ---- linear head: 1x1 convs through the same WMMA path ----
  {
    // lin1: 2048 -> 256, bias + ReLU, bf16 out (act_in == b1 after L14)
    const int P = B * 2 * 2;
    long waves = ((long)P >> 5) * (256 / 32);
    conv_wmma_kernel<1, 2, false, true><<<cdiv(waves * 32, 256), 256, 0, stream>>>(
        act_in, wpkBase + wpkElems, nullptr, nullptr, b2, nullptr, lin_b1,
        B, 2, 2, 2048, 2, 2, 256, 1, 1, 1, 1, 0);
    // lin2: 256 -> 128, bias, fp32 -> d_out
    waves = ((long)P >> 5) * (128 / 32);
    conv_wmma_kernel<2, 2, false, true><<<cdiv(waves * 32, 256), 256, 0, stream>>>(
        b2, wpkBase + wpkElems + lin1Elems, nullptr, nullptr, nullptr,
        (float*)d_out, lin_b2,
        B, 2, 2, 256, 2, 2, 128, 1, 1, 1, 1, 0);
  }
}
